// Proxy_NCA_6124623364210
// MI455X (gfx1250) — matrix operand order repaired and checked by reading, not compiled, not run on hardware
//
#include <hip/hip_runtime.h>
#include <hip/hip_bf16.h>

typedef _Float16 __attribute__((ext_vector_type(16))) v16h;
typedef _Float16 __attribute__((ext_vector_type(8)))  v8h;
typedef float    __attribute__((ext_vector_type(8)))  v8f;

#define DK      512
#define B_ROWS  1024
#define C_COLS  32768

// ---------------------------------------------------------------------------
// Normalize rows to norm 3, emit f16 hi/lo split + squared norm (post-scale).
// One wave (32 lanes) per row of 512 floats; 16 elements per lane.
// ---------------------------------------------------------------------------
__global__ __launch_bounds__(128)
void normalize_split_kernel(const float* __restrict__ src, int rows,
                            _Float16* __restrict__ hi, _Float16* __restrict__ lo,
                            float* __restrict__ sq) {
    const int wave = (blockIdx.x * blockDim.x + threadIdx.x) >> 5;
    const int lane = threadIdx.x & 31;
    if (wave >= rows) return;

    const float* r = src + (size_t)wave * DK;
    float x[16];
    float ss = 0.f;
#pragma unroll
    for (int j = 0; j < 16; ++j) {
        x[j] = r[lane + 32 * j];
        ss += x[j] * x[j];
    }
#pragma unroll
    for (int off = 16; off >= 1; off >>= 1) ss += __shfl_xor(ss, off, 32);

    const float n = sqrtf(ss);
    const float s = 3.0f / fmaxf(n, 1e-12f);

    _Float16* hr = hi + (size_t)wave * DK;
    _Float16* lr = lo + (size_t)wave * DK;
    float bs = 0.f;
#pragma unroll
    for (int j = 0; j < 16; ++j) {
        const float v = x[j] * s;
        bs += v * v;
        const _Float16 h = (_Float16)v;
        hr[lane + 32 * j] = h;
        lr[lane + 32 * j] = (_Float16)(v - (float)h);
    }
#pragma unroll
    for (int off = 16; off >= 1; off >>= 1) bs += __shfl_xor(bs, off, 32);
    if (lane == 0) sq[wave] = bs;
}

__global__ void zero_kernel(float* __restrict__ p, int n) {
    const int i = blockIdx.x * blockDim.x + threadIdx.x;
    if (i < n) p[i] = 0.f;
}

// ---------------------------------------------------------------------------
// Fused split-f16 WMMA GEMM (S = batch @ P^T) + masked sum of exp(-dist).
// Block = 128 threads (4 waves). Block tile: M=32 rows x N=256 cols.
// Each wave: 2 M-tiles x 4 N-tiles of 16x16; every B fragment feeds 6 WMMAs.
// K=512 processed in 2 phases of 256 (A hi/lo staged in 32 KB LDS per phase),
// 8 chunks of 32 per phase; 3 WMMAs per chunk per tile (hh, hl, lh).
// ---------------------------------------------------------------------------
__global__ __launch_bounds__(128)
void gemm_lse_kernel(const _Float16* __restrict__ Ahi, const _Float16* __restrict__ Alo,
                     const _Float16* __restrict__ Bhi, const _Float16* __restrict__ Blo,
                     const float* __restrict__ bsq, const float* __restrict__ psq,
                     const int* __restrict__ labels, float* __restrict__ rowsum) {
    __shared__ alignas(16) _Float16 Ah_s[32 * 256];   // 16 KB
    __shared__ alignas(16) _Float16 Al_s[32 * 256];   // 16 KB
    __shared__ float bsq_s[32];
    __shared__ int   lab_s[32];

    const int tid = threadIdx.x;
    const int m0  = blockIdx.y * 32;
    const int c0  = blockIdx.x * 256;

    if (tid < 32) {
        bsq_s[tid] = bsq[m0 + tid];
        lab_s[tid] = labels[m0 + tid];
    }

    const int wid  = tid >> 5;
    const int lane = tid & 31;
    const int l16  = lane & 15;   // A: M row within tile / B,C: N col within tile
    const int half = lane >> 4;   // lane-half: K interleave select / M offset
    const int cw   = c0 + wid * 64;

    v8f acc[2][4] = {};

    for (int phase = 0; phase < 2; ++phase) {
        __syncthreads();   // previous-phase readers done before overwrite
        // Stage A[m0..m0+31][phase*256 .. +256) hi/lo: 1024 float4 per split.
        {
            float4* sh = (float4*)Ah_s;
            float4* sl = (float4*)Al_s;
#pragma unroll
            for (int i = 0; i < 8; ++i) {
                const int c   = tid + 128 * i;
                const int row = c >> 5;
                const int cc  = c & 31;
                const int gidx = (m0 + row) * 64 + phase * 32 + cc; // float4 units
                sh[row * 32 + cc] = ((const float4*)Ahi)[gidx];
                sl[row * 32 + cc] = ((const float4*)Alo)[gidx];
            }
        }
        __syncthreads();

        for (int kc = 0; kc < 8; ++kc) {
            const int kl = kc * 32;              // K offset within LDS stage
            const int k0 = phase * 256 + kl;     // global K offset

            // 16-bit A layout: lane half 0 holds K k0..k0+15,
            // lane half 1 holds K k0+16..k0+31 (matches B).
            union { v16h v; v8h h[2]; } a_hi[2], a_lo[2];
#pragma unroll
            for (int mt = 0; mt < 2; ++mt) {
                const int abase = (l16 + 16 * mt) * 256 + kl + 16 * half;
                a_hi[mt].h[0] = *(const v8h*)&Ah_s[abase];
                a_hi[mt].h[1] = *(const v8h*)&Ah_s[abase + 8];
                a_lo[mt].h[0] = *(const v8h*)&Al_s[abase];
                a_lo[mt].h[1] = *(const v8h*)&Al_s[abase + 8];
            }

#pragma unroll
            for (int t = 0; t < 4; ++t) {
                const int col = cw + t * 16 + l16;
                // 16-bit B layout: lanes 0-15 hold K k0..k0+15, lanes 16-31
                // hold K k0+16..k0+31 -> 32 contiguous bytes of a proxy row.
                // 32-bit offset => SGPR-base + VGPR-offset addressing.
                const unsigned bofs = (unsigned)col * DK + (unsigned)(k0 + 16 * half);
                const v16h bh = *(const v16h*)(Bhi + bofs);
                const v16h bl = *(const v16h*)(Blo + bofs);
                __builtin_prefetch(Bhi + bofs + 128, 0, 3); // ahead along K
#pragma unroll
                for (int mt = 0; mt < 2; ++mt) {
                    acc[mt][t] = __builtin_amdgcn_wmma_f32_16x16x32_f16(
                        false, a_hi[mt].v, false, bh, (short)0, acc[mt][t], false, false);
                    acc[mt][t] = __builtin_amdgcn_wmma_f32_16x16x32_f16(
                        false, a_hi[mt].v, false, bl, (short)0, acc[mt][t], false, false);
                    acc[mt][t] = __builtin_amdgcn_wmma_f32_16x16x32_f16(
                        false, a_lo[mt].v, false, bh, (short)0, acc[mt][t], false, false);
                }
            }
        }
    }

    // Epilogue: dist = bsq + psq - 2S >= 0, so exp(-dist) <= 1 (no overflow).
    // C/D layout: VGPR r, lanes 0-15 -> M=r, lanes 16-31 -> M=r+8; N = lane&15.
    float rowterm[2][8];
#pragma unroll
    for (int mt = 0; mt < 2; ++mt)
#pragma unroll
        for (int r = 0; r < 8; ++r) rowterm[mt][r] = 0.f;

#pragma unroll
    for (int t = 0; t < 4; ++t) {
        const int col = cw + t * 16 + l16;
        const float pc = psq[col];
#pragma unroll
        for (int mt = 0; mt < 2; ++mt) {
#pragma unroll
            for (int r = 0; r < 8; ++r) {
                const int mloc = 16 * mt + r + 8 * half;
                float term = expf(2.f * acc[mt][t][r] - bsq_s[mloc] - pc);
                if (col == lab_s[mloc]) term = 0.f;   // mask positive class
                rowterm[mt][r] += term;
            }
        }
    }
#pragma unroll
    for (int mt = 0; mt < 2; ++mt) {
#pragma unroll
        for (int r = 0; r < 8; ++r) {
            float v = rowterm[mt][r];
            v += __shfl_xor(v, 1, 32);
            v += __shfl_xor(v, 2, 32);
            v += __shfl_xor(v, 4, 32);
            v += __shfl_xor(v, 8, 32);    // sum within each 16-lane half
            if (l16 == 0)
                atomicAdd(&rowsum[m0 + 16 * mt + r + 8 * half], v);
        }
    }
}

// ---------------------------------------------------------------------------
// Finalize: fp32 positive distance (from hi+lo splits, fp32 accumulate),
// + log(rowsum), deterministic block reduction to the scalar mean.
// ---------------------------------------------------------------------------
__global__ __launch_bounds__(1024)
void finalize_kernel(const _Float16* __restrict__ Ahi, const _Float16* __restrict__ Alo,
                     const _Float16* __restrict__ Bhi, const _Float16* __restrict__ Blo,
                     const float* __restrict__ bsq, const float* __restrict__ psq,
                     const int* __restrict__ labels, const float* __restrict__ rowsum,
                     float* __restrict__ out) {
    __shared__ float red[1024];
    const int b   = threadIdx.x;
    const int lab = labels[b];
    const _Float16* ah = Ahi + (size_t)b * DK;
    const _Float16* al = Alo + (size_t)b * DK;
    const _Float16* bh = Bhi + (size_t)lab * DK;
    const _Float16* bl = Blo + (size_t)lab * DK;
    float dot = 0.f;
    for (int k = 0; k < DK; ++k)
        dot += ((float)ah[k] + (float)al[k]) * ((float)bh[k] + (float)bl[k]);
    const float pos = bsq[b] + psq[lab] - 2.f * dot;
    red[b] = pos + logf(rowsum[b]);
    __syncthreads();
    for (int s = 512; s > 0; s >>= 1) {
        if (b < s) red[b] += red[b + s];
        __syncthreads();
    }
    if (b == 0) out[0] = red[0] * (1.0f / 1024.0f);
}

// ---------------------------------------------------------------------------
extern "C" void kernel_launch(void* const* d_in, const int* in_sizes, int n_in,
                              void* d_out, int out_size, void* d_ws, size_t ws_size,
                              hipStream_t stream) {
    const float* emb    = (const float*)d_in[0];
    const float* prox   = (const float*)d_in[1];
    const int*   labels = (const int*)d_in[2];
    float*       out    = (float*)d_out;

    char* ws = (char*)d_ws;
    size_t off = 0;
    auto carve = [&](size_t bytes) -> void* {
        void* p = ws + off;
        off = (off + bytes + 255) & ~(size_t)255;
        return p;
    };
    _Float16* Ahi = (_Float16*)carve((size_t)B_ROWS * DK * sizeof(_Float16));
    _Float16* Alo = (_Float16*)carve((size_t)B_ROWS * DK * sizeof(_Float16));
    _Float16* Bhi = (_Float16*)carve((size_t)C_COLS * DK * sizeof(_Float16));
    _Float16* Blo = (_Float16*)carve((size_t)C_COLS * DK * sizeof(_Float16));
    float*    bsq = (float*)carve((size_t)B_ROWS * sizeof(float));
    float*    psq = (float*)carve((size_t)C_COLS * sizeof(float));
    float*    rowsum = (float*)carve((size_t)B_ROWS * sizeof(float));

    normalize_split_kernel<<<B_ROWS / 4, 128, 0, stream>>>(emb, B_ROWS, Ahi, Alo, bsq);
    normalize_split_kernel<<<C_COLS / 4, 128, 0, stream>>>(prox, C_COLS, Bhi, Blo, psq);
    zero_kernel<<<(B_ROWS + 255) / 256, 256, 0, stream>>>(rowsum, B_ROWS);

    dim3 grid(C_COLS / 256, B_ROWS / 32);   // (128, 32)
    gemm_lse_kernel<<<grid, 128, 0, stream>>>(Ahi, Alo, Bhi, Blo, bsq, psq, labels, rowsum);

    finalize_kernel<<<1, 1024, 0, stream>>>(Ahi, Alo, Bhi, Blo, bsq, psq, labels, rowsum, out);
}
